// CMM_44332652430101
// MI455X (gfx1250) — matrix operand order repaired and checked by reading, not compiled, not run on hardware
//
#include <hip/hip_runtime.h>

// ---------------------------------------------------------------------------
// Problem constants (match reference)
#define BB   32
#define FF   16
#define NN   197
#define EDIM 768
#define PDIM 256
#define SZ   64
#define KSEL 8
#define BZ   (BB * FF)        // 512

typedef __bf16 bf16_t;
typedef __attribute__((ext_vector_type(16))) __bf16 v16bf;
typedef __attribute__((ext_vector_type(8)))  __bf16 v8bf;
typedef __attribute__((ext_vector_type(8)))  float  v8f;

// ---------------------------------------------------------------------------
// bf16 split helpers (manual RNE so we never depend on __bf16 cast support)
__device__ __forceinline__ unsigned short f32_to_bf16_bits(float f) {
    unsigned int u = __float_as_uint(f);
    u += 0x7FFFu + ((u >> 16) & 1u);
    return (unsigned short)(u >> 16);
}
__device__ __forceinline__ float bf16_bits_to_f32(unsigned short h) {
    return __uint_as_float(((unsigned int)h) << 16);
}
__device__ __forceinline__ void store_hilo(unsigned short* hi, unsigned short* lo,
                                           size_t o, float f) {
    unsigned short hb = f32_to_bf16_bits(f);
    hi[o] = hb;
    lo[o] = f32_to_bf16_bits(f - bf16_bits_to_f32(hb));
}

// 256-thread block sum; returns total to all threads, safe to call repeatedly
__device__ __forceinline__ float block_sum_256(float v, float* sh) {
    int t = threadIdx.x;
    __syncthreads();
    sh[t] = v;
    __syncthreads();
    for (int s = 128; s > 0; s >>= 1) {
        if (t < s) sh[t] += sh[t + s];
        __syncthreads();
    }
    float r = sh[0];
    __syncthreads();
    return r;
}

// ---------------------------------------------------------------------------
// Stage 0: split fp32 weights into bf16 hi/lo
__global__ void cmm_split_bf16(const float* __restrict__ src,
                               unsigned short* __restrict__ hi,
                               unsigned short* __restrict__ lo, int n) {
    int i = blockIdx.x * blockDim.x + threadIdx.x;
    if (i < n) store_hilo(hi, lo, i, src[i]);
}

// ---------------------------------------------------------------------------
// Stage 1: frame mean  xf[bz, e] = mean_n x[bz*197 + n, e]   (streams 310 MB)
// One block per bz row; 192 threads x float4 = 768 floats per row.
__global__ __launch_bounds__(192)
void cmm_frame_mean(const float* __restrict__ x,
                    unsigned short* __restrict__ xf_hi,
                    unsigned short* __restrict__ xf_lo) {
    int bz = blockIdx.x;
    int t  = threadIdx.x;                 // 0..191
    const float* base = x + (size_t)bz * NN * EDIM + t * 4;
    float a0 = 0.f, a1 = 0.f, a2 = 0.f, a3 = 0.f;
    for (int n = 0; n < NN; ++n) {
        float4 v = *(const float4*)(base + (size_t)n * EDIM);
        a0 += v.x; a1 += v.y; a2 += v.z; a3 += v.w;
    }
    const float inv = 1.0f / (float)NN;
    size_t o = (size_t)bz * EDIM + t * 4;
    store_hilo(xf_hi, xf_lo, o + 0, a0 * inv);
    store_hilo(xf_hi, xf_lo, o + 1, a1 * inv);
    store_hilo(xf_hi, xf_lo, o + 2, a2 * inv);
    store_hilo(xf_hi, xf_lo, o + 3, a3 * inv);
}

// ---------------------------------------------------------------------------
// Stage 2: keys = l2norm(prompt_values[:,0,:])    (64 rows x 256)
__global__ __launch_bounds__(256)
void cmm_keys_norm(const float* __restrict__ pv, float* __restrict__ keys) {
    __shared__ float sh[256];
    int s = blockIdx.x, t = threadIdx.x;
    float v  = pv[s * PDIM + t];
    float ss = block_sum_256(v * v, sh);
    keys[s * PDIM + t] = v / fmaxf(sqrtf(ss), 1e-12f);
}

// Stage 2b: ksim_sum = sum |keys keys^T - I|  ; also zero diff accumulator
__global__ __launch_bounds__(256)
void cmm_ksim(const float* __restrict__ keys, float* __restrict__ scal) {
    __shared__ float sh[256];
    int t = threadIdx.x;
    float acc = 0.f;
    for (int pair = t; pair < SZ * SZ; pair += 256) {
        int i = pair >> 6, j = pair & 63;
        const float* ki = keys + i * PDIM;
        const float* kj = keys + j * PDIM;
        float d = 0.f;
        for (int p = 0; p < PDIM; ++p) d += ki[p] * kj[p];
        acc += fabsf(d - (i == j ? 1.0f : 0.0f));
    }
    float total = block_sum_256(acc, sh);
    if (t == 0) { scal[1] = total; scal[0] = 0.0f; }   // scal[0] = diff accum
}

// ---------------------------------------------------------------------------
// WMMA GEMM, bf16x3 split for ~fp32 accuracy.
//   D[M,N] (f32) = A[M,K] * B^T   where Bstore is [N,K] row-major.
// One wave32 per 16x16 output tile. Fragment layouts follow CDNA5 ISA 7.12.2.
__device__ __forceinline__ v16bf load_a_frag(const bf16_t* __restrict__ A,
                                             int m, int Kd, int k0, int hb) {
    // lane<16: K = k0+{0..7, 16..23}; lane>=16: K = k0+{8..15, 24..31}
    const bf16_t* p = A + (size_t)m * Kd + k0 + 8 * hb;
    v8bf a0 = *(const v8bf*)p;
    v8bf a1 = *(const v8bf*)(p + 16);
    v16bf r;
#pragma unroll
    for (int i = 0; i < 8; ++i) { r[i] = a0[i]; r[i + 8] = a1[i]; }
    return r;
}
__device__ __forceinline__ v16bf load_b_frag(const bf16_t* __restrict__ Bs,
                                             int n, int Kd, int k0, int hb) {
    // lane<16: K = k0+{0..7, 16..23} of column n; lane>=16: K = k0+{8..15, 24..31}
    const bf16_t* p = Bs + (size_t)n * Kd + k0 + 8 * hb;
    v8bf b0 = *(const v8bf*)p;
    v8bf b1 = *(const v8bf*)(p + 16);
    v16bf r;
#pragma unroll
    for (int i = 0; i < 8; ++i) { r[i] = b0[i]; r[i + 8] = b1[i]; }
    return r;
}

__global__ __launch_bounds__(256)
void cmm_gemm_bf16x3(const bf16_t* __restrict__ Ahi, const bf16_t* __restrict__ Alo,
                     const bf16_t* __restrict__ Bhi, const bf16_t* __restrict__ Blo,
                     float* __restrict__ D, int M, int Nd, int Kd) {
    int wavesPerBlock = blockDim.x >> 5;
    int wave   = threadIdx.x >> 5;
    int tile   = blockIdx.x * wavesPerBlock + wave;
    int tilesN = Nd >> 4;
    int tm = tile / tilesN;
    int tn = tile - tm * tilesN;
    if (tm * 16 >= M) return;              // wave-uniform guard (EXEC stays full)
    int lane = threadIdx.x & 31;
    int l16  = lane & 15;
    int hb   = lane >> 4;
    int m = tm * 16 + l16;
    int n = tn * 16 + l16;
    v8f c = {};
    for (int k0 = 0; k0 < Kd; k0 += 32) {
        v16bf ah = load_a_frag(Ahi, m, Kd, k0, hb);
        v16bf al = load_a_frag(Alo, m, Kd, k0, hb);
        v16bf bh = load_b_frag(Bhi, n, Kd, k0, hb);
        v16bf bl = load_b_frag(Blo, n, Kd, k0, hb);
        c = __builtin_amdgcn_wmma_f32_16x16x32_bf16(false, ah, false, bh, (short)0, c, false, false);
        c = __builtin_amdgcn_wmma_f32_16x16x32_bf16(false, ah, false, bl, (short)0, c, false, false);
        c = __builtin_amdgcn_wmma_f32_16x16x32_bf16(false, al, false, bh, (short)0, c, false, false);
    }
#pragma unroll
    for (int r = 0; r < 8; ++r) {
        int row = tm * 16 + r + 8 * hb;    // C/D layout: VGPR r -> M=r (+8 for hi half)
        D[(size_t)row * Nd + tn * 16 + l16] = c[r];
    }
}

// ---------------------------------------------------------------------------
// Stage 4: per-row normalize, sim, top-8, recon/diff, normalized prompts
__global__ __launch_bounds__(256)
void cmm_route(const float* __restrict__ q_raw, const float* __restrict__ keys,
               const float* __restrict__ pv,
               unsigned short* __restrict__ p_hi, unsigned short* __restrict__ p_lo,
               float* __restrict__ scal) {
    __shared__ float sh[256];
    __shared__ float q_sh[PDIM];
    __shared__ float sim_sh[SZ];
    __shared__ int   idx_sh[KSEL];
    __shared__ float ssim_sh[KSEL];

    int bz = blockIdx.x, t = threadIdx.x;

    // q = l2norm(q_raw row)
    float qa = q_raw[(size_t)bz * PDIM + t];
    float ss = block_sum_256(qa * qa, sh);
    float qn = qa / fmaxf(sqrtf(ss), 1e-12f);
    q_sh[t] = qn;
    __syncthreads();

    // sim[s] = <q, keys[s]> ; 4 threads per s
    {
        int s = t >> 2, part = t & 3;
        const float* kr = keys + s * PDIM + part * 64;
        const float* qr = q_sh + part * 64;
        float p = 0.f;
#pragma unroll 8
        for (int j = 0; j < 64; ++j) p += qr[j] * kr[j];
        sh[t] = p;
        __syncthreads();
        if (part == 0) sim_sh[s] = sh[t] + sh[t + 1] + sh[t + 2] + sh[t + 3];
        __syncthreads();
    }

    // top-8 (first-max tie behavior matches lax.top_k)
    if (t == 0) {
        unsigned long long used = 0ull;
        for (int k = 0; k < KSEL; ++k) {
            float best = -3.4e38f; int bi = 0;
            for (int s2 = 0; s2 < SZ; ++s2) {
                if (used & (1ull << s2)) continue;
                float v = sim_sh[s2];
                if (v > best) { best = v; bi = s2; }
            }
            used |= 1ull << bi;
            idx_sh[k] = bi; ssim_sh[k] = best;
        }
    }
    __syncthreads();

    // recon & diff contribution
    float recon = 0.f;
#pragma unroll
    for (int k = 0; k < KSEL; ++k)
        recon += ssim_sh[k] * keys[idx_sh[k] * PDIM + t];
    float dd = recon - qn;
    float dsum = block_sum_256(dd * dd, sh);
    if (t == 0) atomicAdd(&scal[0], dsum);

    // normalized selected prompts -> bf16 hi/lo for the out-proj GEMM
    for (int k = 0; k < KSEL; ++k) {
        float v  = pv[idx_sh[k] * PDIM + t];
        float sq = block_sum_256(v * v, sh);
        float pn = v / fmaxf(sqrtf(sq), 1e-12f);
        size_t o = ((size_t)bz * KSEL + k) * PDIM + t;   // row = b*128 + f*8 + k
        store_hilo(p_hi, p_lo, o, pn);
    }
}

// ---------------------------------------------------------------------------
// Stage 6: ps_loss = ((diff_sum + ksim_sum)/BZ) * F
__global__ void cmm_finalize(const float* __restrict__ scal, float* __restrict__ out_loss) {
    out_loss[0] = (scal[0] + scal[1]) * ((float)FF / (float)BZ);
}

// ---------------------------------------------------------------------------
extern "C" void kernel_launch(void* const* d_in, const int* in_sizes, int n_in,
                              void* d_out, int out_size, void* d_ws, size_t ws_size,
                              hipStream_t stream) {
    (void)in_sizes; (void)n_in; (void)out_size; (void)ws_size;
    const float* x     = (const float*)d_in[0];   // [32, 3152, 768]
    const float* W_in  = (const float*)d_in[1];   // [256, 768]
    const float* W_out = (const float*)d_in[2];   // [768, 256]
    const float* pv    = (const float*)d_in[3];   // [64, 1, 256]
    float* out = (float*)d_out;                   // 32*128*768 floats + 1 loss

    // workspace carving (256B aligned)
    size_t off = 0;
    char* base = (char*)d_ws;
    auto take = [&](size_t bytes) -> void* {
        void* p = base + off;
        off += (bytes + 255) & ~(size_t)255;
        return p;
    };
    unsigned short* xf_hi   = (unsigned short*)take((size_t)BZ * EDIM * 2);
    unsigned short* xf_lo   = (unsigned short*)take((size_t)BZ * EDIM * 2);
    unsigned short* win_hi  = (unsigned short*)take((size_t)PDIM * EDIM * 2);
    unsigned short* win_lo  = (unsigned short*)take((size_t)PDIM * EDIM * 2);
    unsigned short* wout_hi = (unsigned short*)take((size_t)EDIM * PDIM * 2);
    unsigned short* wout_lo = (unsigned short*)take((size_t)EDIM * PDIM * 2);
    float*          q_raw   = (float*)take((size_t)BZ * PDIM * 4);
    float*          keys    = (float*)take((size_t)SZ * PDIM * 4);
    unsigned short* p_hi    = (unsigned short*)take((size_t)BZ * KSEL * PDIM * 2);
    unsigned short* p_lo    = (unsigned short*)take((size_t)BZ * KSEL * PDIM * 2);
    float*          scal    = (float*)take(256);   // [0]=diff_sum, [1]=ksim_sum

    // Stage 0: weight splits
    cmm_split_bf16<<<(PDIM * EDIM + 255) / 256, 256, 0, stream>>>(W_in,  win_hi,  win_lo,  PDIM * EDIM);
    cmm_split_bf16<<<(EDIM * PDIM + 255) / 256, 256, 0, stream>>>(W_out, wout_hi, wout_lo, EDIM * PDIM);

    // Stage 1: frame mean (bandwidth-dominant, ~310 MB streamed)
    cmm_frame_mean<<<BZ, 192, 0, stream>>>(x, xf_hi, xf_lo);

    // Stage 2: keys + ksim (+ zero diff accumulator)
    cmm_keys_norm<<<SZ, 256, 0, stream>>>(pv, keys);
    cmm_ksim<<<1, 256, 0, stream>>>(keys, scal);

    // Stage 3: q_raw = xf @ W_in^T   (M=512, N=256, K=768) -> 512 tiles / 8 waves
    cmm_gemm_bf16x3<<<(BZ / 16) * (PDIM / 16) / 8, 256, 0, stream>>>(
        (const bf16_t*)xf_hi, (const bf16_t*)xf_lo,
        (const bf16_t*)win_hi, (const bf16_t*)win_lo,
        q_raw, BZ, PDIM, EDIM);

    // Stage 4: normalize / sim / top-8 / losses / prompt prep
    cmm_route<<<BZ, 256, 0, stream>>>(q_raw, keys, pv, p_hi, p_lo, scal);

    // Stage 5: out = prompts @ W_out^T  (M=4096, N=768, K=256) -> 12288 tiles
    cmm_gemm_bf16x3<<<((BZ * KSEL) / 16) * (EDIM / 16) / 8, 256, 0, stream>>>(
        (const bf16_t*)p_hi, (const bf16_t*)p_lo,
        (const bf16_t*)wout_hi, (const bf16_t*)wout_lo,
        out, BZ * KSEL, EDIM, PDIM);

    // Stage 6: ps_loss scalar appended after out_prompts
    cmm_finalize<<<1, 1, 0, stream>>>(scal, out + (size_t)BB * FF * KSEL * EDIM);
}
